// NodeAttention_66795331387432
// MI455X (gfx1250) — matrix operand order repaired and checked by reading, not compiled, not run on hardware
//
#include <hip/hip_runtime.h>
#include <hip/hip_bf16.h>

typedef __attribute__((ext_vector_type(16))) __bf16 v16bf;
typedef __attribute__((ext_vector_type(8)))  __bf16 v8bf;
typedef __attribute__((ext_vector_type(8)))  float  v8f;
typedef __attribute__((ext_vector_type(4)))  unsigned int uint4v;
typedef __attribute__((ext_vector_type(8)))  unsigned int uint8v;

constexpr int kBT = 16;        // B*T
constexpr int kN  = 512;       // nodes
constexpr int kE  = 768;       // embed
constexpr int kH  = 12;        // heads
constexpr int kHd = 64;        // head dim
constexpr int kE3 = 2304;      // 3*E

// ---------------------------------------------------------------------------
// WMMA fragment helpers (gfx1250, wave32)
// ---------------------------------------------------------------------------
static __device__ inline v8f wmma_bf16(v16bf a, v16bf b, v8f c) {
  return __builtin_amdgcn_wmma_f32_16x16x32_bf16(false, a, false, b, (short)0, c,
                                                 false, false);
}

static __device__ inline v16bf b_frag(const __bf16* p) {
  return *reinterpret_cast<const v16bf*>(p);
}

static __device__ inline v16bf a_frag_bf16(const __bf16* p, int sel) {
  v8bf lo = *reinterpret_cast<const v8bf*>(p + sel * 16);
  v8bf hi = *reinterpret_cast<const v8bf*>(p + sel * 16 + 8);
  v16bf a;
#pragma unroll
  for (int i = 0; i < 8; ++i) { a[i] = lo[i]; a[i + 8] = hi[i]; }
  return a;
}

static __device__ inline v16bf a_frag_f32(const float* p, int sel) {
  float t[16] __attribute__((aligned(16)));
  *reinterpret_cast<float4*>(&t[0])  = *reinterpret_cast<const float4*>(p + sel * 16);
  *reinterpret_cast<float4*>(&t[4])  = *reinterpret_cast<const float4*>(p + sel * 16 + 4);
  *reinterpret_cast<float4*>(&t[8])  = *reinterpret_cast<const float4*>(p + sel * 16 + 8);
  *reinterpret_cast<float4*>(&t[12]) = *reinterpret_cast<const float4*>(p + sel * 16 + 12);
  v16bf a;
#pragma unroll
  for (int i = 0; i < 16; ++i) a[i] = (__bf16)t[i];
  return a;
}

// ---------------------------------------------------------------------------
// Tensor Data Mover: 1-D bf16 copy global -> LDS via D# in SGPRs.
// D# layout per CDNA5 ISA 8.3/8.4: group0 = {count/type/addrs}, group1 = dims.
// ---------------------------------------------------------------------------
static __device__ inline void tdm_copy_1d(const void* gsrc, void* ldst,
                                          unsigned nelem /* 2-byte elems */) {
  const unsigned long long ga = (unsigned long long)(uintptr_t)gsrc;
  uint4v g0;
  g0[0] = 1u;                                   // count=1, user descriptor
  g0[1] = (unsigned)(uintptr_t)ldst;            // lds_addr (low 32 = LDS offset)
  g0[2] = (unsigned)(ga & 0xffffffffu);         // global_addr[31:0]
  g0[3] = (unsigned)((ga >> 32) & 0x1ffffffu)   // global_addr[56:32]
          | 0x80000000u;                        // type=2 ("image")
  uint8v g1;
  g1[0] = (1u << 16);                           // wg_mask=0, data_size=1 (2B)
  g1[1] = (nelem & 0xffffu) << 16;              // tensor_dim0[15:0] @ bit48
  g1[2] = (nelem >> 16) & 0xffffu;              // tensor_dim0[31:16]; tensor_dim1=0
  g1[3] = (nelem & 0xffffu) << 16;              // tile_dim0 @ bit112
  g1[4] = 0u;                                   // tile_dim1=0, tile_dim2=0
  g1[5] = nelem;                                // tensor_dim0_stride[31:0]
  g1[6] = 0u;
  g1[7] = 0u;
  // Drain our own LDS reads before the DMA may overwrite the buffer (TDM is
  // unordered vs other memory instruction types).
  asm volatile("s_wait_dscnt 0x0" ::: "memory");
  asm volatile("tensor_load_to_lds %0, %1" :: "s"(g0), "s"(g1) : "memory");
}

// ---------------------------------------------------------------------------
// Kernel 1: QKV projection. grid=(2304/64, 8192/64), block=128 (4 waves).
// ---------------------------------------------------------------------------
__global__ __launch_bounds__(128)
void qkv_proj_kernel(const float* __restrict__ hidden,
                     const float* __restrict__ w_attn,
                     const float* __restrict__ b_attn,
                     __bf16* __restrict__ qs, __bf16* __restrict__ ks,
                     __bf16* __restrict__ vt) {
  __shared__ __attribute__((aligned(32))) __bf16 ldsW[64 * 32];  // [col][k]
  const int lane = threadIdx.x & 31;
  const int wave = threadIdx.x >> 5;
  const int sel  = lane >> 4;
  const int ln   = lane & 15;
  const int colbase = blockIdx.x * 64;
  const int rbase   = blockIdx.y * 64 + wave * 16;

  v8f acc[4];
#pragma unroll
  for (int ct = 0; ct < 4; ++ct)
#pragma unroll
    for (int i = 0; i < 8; ++i) acc[ct][i] = 0.f;

  const int kW   = threadIdx.x >> 2;         // 0..31
  const int cseg = (threadIdx.x & 3) * 16;

  for (int kt = 0; kt < 24; ++kt) {
    const int kk = kt * 32;
    const float* wr = w_attn + (size_t)(kk + kW) * kE3 + colbase + cseg;
    float tv[16] __attribute__((aligned(16)));
    *reinterpret_cast<float4*>(&tv[0])  = *reinterpret_cast<const float4*>(wr + 0);
    *reinterpret_cast<float4*>(&tv[4])  = *reinterpret_cast<const float4*>(wr + 4);
    *reinterpret_cast<float4*>(&tv[8])  = *reinterpret_cast<const float4*>(wr + 8);
    *reinterpret_cast<float4*>(&tv[12]) = *reinterpret_cast<const float4*>(wr + 12);
#pragma unroll
    for (int j = 0; j < 16; ++j) ldsW[(cseg + j) * 32 + kW] = (__bf16)tv[j];
    __syncthreads();

    const int arow = rbase + ln;
    v16bf a = a_frag_f32(hidden + (size_t)arow * kE + kk, sel);
#pragma unroll
    for (int ct = 0; ct < 4; ++ct) {
      v16bf b = b_frag(&ldsW[(ct * 16 + ln) * 32 + sel * 16]);
      acc[ct] = wmma_bf16(a, b, acc[ct]);
    }
    __syncthreads();
  }

  // 64-col block lies entirely inside one of Q|K|V -> block-uniform region.
  const int region = colbase / kE;
  const int cqbase = colbase - region * kE;
#pragma unroll
  for (int ct = 0; ct < 4; ++ct) {
    const int col  = colbase + ct * 16 + ln;
    const float bias = b_attn[col];
    const int cq = cqbase + ct * 16 + ln;
    const int h = cq >> 6, d = cq & 63;
#pragma unroll
    for (int r = 0; r < 8; ++r) {
      const int row = rbase + r + sel * 8;
      const int bt = row >> 9, n = row & 511;
      const float v = acc[ct][r] + bias;
      const size_t bh = (size_t)(bt * kH + h);
      if (region == 0)      qs[(bh * kN + n) * kHd + d] = (__bf16)(v * 0.125f);
      else if (region == 1) ks[(bh * kN + n) * kHd + d] = (__bf16)v;
      else                  vt[(bh * kHd + d) * kN + n] = (__bf16)v;
    }
  }
}

// ---------------------------------------------------------------------------
// Kernel 2: fused attention. grid=(512/16, 16*12), block=32 (one wave).
// K/V operand tiles DMA'd into LDS by the Tensor Data Mover (double-buffered
// 8 KB K-chunks in phase A, 16 KB V^T slabs in phase C).
// ---------------------------------------------------------------------------
__global__ __launch_bounds__(32)
void attn_kernel(const __bf16* __restrict__ qs, const __bf16* __restrict__ ks,
                 const __bf16* __restrict__ vt, const float* __restrict__ edge,
                 const float* __restrict__ mask, __bf16* __restrict__ aout) {
  __shared__ __attribute__((aligned(32))) float ldsS[16 * kN];       // 32 KB
  __shared__ __attribute__((aligned(64))) __bf16 ldsKV[2][64 * 64];  // 16 KB
  __shared__ float ldsInv[16];
  const int lane = threadIdx.x & 31;
  const int sel  = lane >> 4;
  const int ln   = lane & 15;
  const int qt   = blockIdx.x;
  const int bth  = blockIdx.y;
  const int bt   = bth / kH;
  const int h    = bth - bt * kH;

  const __bf16* qb = qs + (size_t)bth * kN * kHd;
  const __bf16* kb = ks + (size_t)bth * kN * kHd;
  const __bf16* vb = vt + (size_t)bth * kHd * kN;
  const float*  eb = edge + (size_t)bt * kN * kN;
  const float*  mb = mask + (size_t)bt * kN * kN;

  const int qrow = qt * 16 + ln;
  v16bf aq0 = a_frag_bf16(qb + (size_t)qrow * kHd, sel);
  v16bf aq1 = a_frag_bf16(qb + (size_t)qrow * kHd + 32, sel);

  // ---- Phase A: S = Q*K^T + edge + mask -> LDS, K staged by TDM ----
  tdm_copy_1d(kb, &ldsKV[0][0], 64 * kHd);                 // chunk 0
  for (int c = 0; c < 8; ++c) {
    if (c < 7) {
      tdm_copy_1d(kb + (size_t)(c + 1) * 64 * kHd, &ldsKV[(c + 1) & 1][0],
                  64 * kHd);
      __builtin_amdgcn_s_wait_tensorcnt(1);                // chunk c landed
    } else {
      __builtin_amdgcn_s_wait_tensorcnt(0);
    }
    const __bf16* kc = &ldsKV[c & 1][0];
#pragma unroll
    for (int t = 0; t < 4; ++t) {
      const int nb  = c * 64 + t * 16;   // global n base
      const int nbl = t * 16;            // within chunk
      v8f s;
#pragma unroll
      for (int i = 0; i < 8; ++i) s[i] = 0.f;
      s = wmma_bf16(aq0, b_frag(kc + (nbl + ln) * kHd + sel * 16), s);
      s = wmma_bf16(aq1, b_frag(kc + (nbl + ln) * kHd + 32 + sel * 16), s);
#pragma unroll
      for (int r = 0; r < 8; ++r) {
        const int rin = r + sel * 8;
        const size_t eidx = (size_t)(qt * 16 + rin) * kN + nb + ln;
        ldsS[rin * kN + nb + ln] = s[r] + eb[eidx] + mb[eidx];
      }
    }
  }

  // ---- Phase B: softmax (row r handled by lane pair r, r+16) ----
  {
    float* rowp = &ldsS[ln * kN + sel * 256];
    float m = -3.0e38f;
    for (int c = 0; c < 256; ++c) m = fmaxf(m, rowp[c]);
    m = fmaxf(m, __shfl_xor(m, 16, 32));
    float sum = 0.f;
    for (int c = 0; c < 256; ++c) {
      const float e = __expf(rowp[c] - m);
      rowp[c] = e;
      sum += e;
    }
    sum += __shfl_xor(sum, 16, 32);
    if (sel == 0) ldsInv[ln] = 1.0f / sum;
  }

  // ---- Phase C: O = P * V, V^T slabs staged by TDM (16 KB each) ----
  __bf16* vstage = &ldsKV[0][0];
#pragma unroll
  for (int dt = 0; dt < 4; ++dt) {
    tdm_copy_1d(vb + (size_t)dt * 16 * kN, vstage, 16 * kN);
    __builtin_amdgcn_s_wait_tensorcnt(0);
    v8f o;
#pragma unroll
    for (int i = 0; i < 8; ++i) o[i] = 0.f;
    for (int kk = 0; kk < kN; kk += 32) {
      v16bf ap = a_frag_f32(&ldsS[ln * kN + kk], sel);
      v16bf bv = b_frag(vstage + ln * kN + kk + sel * 16);
      o = wmma_bf16(ap, bv, o);
    }
#pragma unroll
    for (int r = 0; r < 8; ++r) {
      const int rin = r + sel * 8;
      const int qg  = qt * 16 + rin;
      const float inv = ldsInv[rin];
      aout[((size_t)(bt * kN) + qg) * kE + h * kHd + dt * 16 + ln] =
          (__bf16)(o[r] * inv);
    }
  }
}

// ---------------------------------------------------------------------------
// Kernel 3: output projection. grid=(768/64, 8192/64), block=128 (4 waves).
// ---------------------------------------------------------------------------
__global__ __launch_bounds__(128)
void out_proj_kernel(const __bf16* __restrict__ x,
                     const float* __restrict__ w_proj,
                     const float* __restrict__ b_proj,
                     float* __restrict__ out) {
  __shared__ __attribute__((aligned(32))) __bf16 ldsW[64 * 32];
  const int lane = threadIdx.x & 31;
  const int wave = threadIdx.x >> 5;
  const int sel  = lane >> 4;
  const int ln   = lane & 15;
  const int colbase = blockIdx.x * 64;
  const int rbase   = blockIdx.y * 64 + wave * 16;

  v8f acc[4];
#pragma unroll
  for (int ct = 0; ct < 4; ++ct)
#pragma unroll
    for (int i = 0; i < 8; ++i) acc[ct][i] = 0.f;

  const int kW   = threadIdx.x >> 2;
  const int cseg = (threadIdx.x & 3) * 16;

  for (int kt = 0; kt < 24; ++kt) {
    const int kk = kt * 32;
    const float* wr = w_proj + (size_t)(kk + kW) * kE + colbase + cseg;
    float tv[16] __attribute__((aligned(16)));
    *reinterpret_cast<float4*>(&tv[0])  = *reinterpret_cast<const float4*>(wr + 0);
    *reinterpret_cast<float4*>(&tv[4])  = *reinterpret_cast<const float4*>(wr + 4);
    *reinterpret_cast<float4*>(&tv[8])  = *reinterpret_cast<const float4*>(wr + 8);
    *reinterpret_cast<float4*>(&tv[12]) = *reinterpret_cast<const float4*>(wr + 12);
#pragma unroll
    for (int j = 0; j < 16; ++j) ldsW[(cseg + j) * 32 + kW] = (__bf16)tv[j];
    __syncthreads();

    const int arow = rbase + ln;
    v16bf a = a_frag_bf16(x + (size_t)arow * kE + kk, sel);
#pragma unroll
    for (int ct = 0; ct < 4; ++ct) {
      v16bf b = b_frag(&ldsW[(ct * 16 + ln) * 32 + sel * 16]);
      acc[ct] = wmma_bf16(a, b, acc[ct]);
    }
    __syncthreads();
  }

#pragma unroll
  for (int ct = 0; ct < 4; ++ct) {
    const int col  = colbase + ct * 16 + ln;
    const float bias = b_proj[col];
#pragma unroll
    for (int r = 0; r < 8; ++r) {
      const int row = rbase + r + sel * 8;
      out[(size_t)row * kE + col] = acc[ct][r] + bias;
    }
  }
}

// ---------------------------------------------------------------------------
extern "C" void kernel_launch(void* const* d_in, const int* in_sizes, int n_in,
                              void* d_out, int out_size, void* d_ws, size_t ws_size,
                              hipStream_t stream) {
  (void)in_sizes; (void)n_in; (void)out_size; (void)ws_size;
  const float* hidden = (const float*)d_in[0];
  const float* edge   = (const float*)d_in[1];
  const float* mask   = (const float*)d_in[2];
  const float* w_attn = (const float*)d_in[3];
  const float* b_attn = (const float*)d_in[4];
  const float* w_proj = (const float*)d_in[5];
  const float* b_proj = (const float*)d_in[6];
  float* out = (float*)d_out;

  const size_t per = (size_t)kBT * kH * kN * kHd;  // 6291456 elements
  char* wsb = (char*)d_ws;
  __bf16* qs = (__bf16*)(wsb);
  __bf16* ks = (__bf16*)(wsb + 2 * per);
  __bf16* vt = (__bf16*)(wsb + 4 * per);
  __bf16* ax = (__bf16*)(wsb + 6 * per);

  qkv_proj_kernel<<<dim3(kE3 / 64, (kBT * kN) / 64), 128, 0, stream>>>(
      hidden, w_attn, b_attn, qs, ks, vt);
  attn_kernel<<<dim3(kN / 16, kBT * kH), 32, 0, stream>>>(
      qs, ks, vt, edge, mask, ax);
  out_proj_kernel<<<dim3(kE / 64, (kBT * kN) / 64), 128, 0, stream>>>(
      ax, w_proj, b_proj, out);
}
